// MHSAUnmasked_901943132414
// MI455X (gfx1250) — matrix operand order repaired and checked by reading, not compiled, not run on hardware
//
#include <hip/hip_runtime.h>

#define BATCH 2
#define TOK   2048
#define DIM   1024
#define NH    16
#define DHEAD 64
#define BT    (BATCH * TOK)   // 4096

typedef __bf16 v16bf __attribute__((ext_vector_type(16)));
typedef float  v8f   __attribute__((ext_vector_type(8)));

union Frag {
    v16bf v;
    uint4 q[2];
};

// native f32 -> bf16 (compiler emits packed cvt where possible)
static __device__ __forceinline__ unsigned short f2bf(float f) {
    return __builtin_bit_cast(unsigned short, (__bf16)f);
}

static __device__ __forceinline__ v8f wmma_bf16(v16bf a, v16bf b, v8f c) {
    // (neg_a, A, neg_b, B, c_mod, C, reuse_a, reuse_b)
    return __builtin_amdgcn_wmma_f32_16x16x32_bf16(false, a, false, b, (short)0, c,
                                                   false, false);
}

// A fragment (16x32, 16-bit): lane `ln` = row, two contiguous 8-elem runs:
//   K = k0 + 8*hi + [0..7]  and  K = k0 + 16 + 8*hi + [0..7]
static __device__ __forceinline__ v16bf load_a(const unsigned short* __restrict__ base,
                                               int row, int k0, int hi, int ld) {
    const unsigned short* r0 = base + (size_t)row * ld + k0 + 8 * hi;
    Frag f;
    f.q[0] = *(const uint4*)r0;
    f.q[1] = *(const uint4*)(r0 + 16);
    return f.v;
}

// B fragment (32x16, 16-bit): lane `ln` = col, two contiguous 8-elem runs:
//   K = k0 + 8*hi + [0..7]  and  K = k0 + 16 + 8*hi + [0..7]   (source is row-major along K with stride ld)
static __device__ __forceinline__ v16bf load_b(const unsigned short* __restrict__ base,
                                               int row, int k0, int hi, int ld) {
    const unsigned short* r0 = base + (size_t)row * ld + k0 + 8 * hi;
    Frag f;
    f.q[0] = *(const uint4*)r0;
    f.q[1] = *(const uint4*)(r0 + 16);
    return f.v;
}

// ---------------------------------------------------------------------------
// Kernel 0: f32 -> bf16 conversion, 8 elements/thread, b128 in / b128 out.
// ---------------------------------------------------------------------------
__global__ __launch_bounds__(256) void cvt_f32_to_bf16(const float* __restrict__ in,
                                                       unsigned short* __restrict__ out,
                                                       int n) {
    const int i = (blockIdx.x * 256 + threadIdx.x) * 8;
    if (i >= n) return;
    float4 f0 = *(const float4*)(in + i);
    float4 f1 = *(const float4*)(in + i + 4);
    unsigned short u[8];
    u[0] = f2bf(f0.x); u[1] = f2bf(f0.y); u[2] = f2bf(f0.z); u[3] = f2bf(f0.w);
    u[4] = f2bf(f1.x); u[5] = f2bf(f1.y); u[6] = f2bf(f1.z); u[7] = f2bf(f1.w);
    *(uint4*)(out + i) = *(const uint4*)u;
}

// ---------------------------------------------------------------------------
// Kernel 1: fused QKV projection. out = x @ W^T + b. One wave per 16x64
// output tile (A fragment reused across 4 WMMAs). Q/K written [B,H,T,DH],
// V written transposed [B,H,DH,T] so the PV B-fragment loads are contiguous.
// grid = (DIM/64, BT/16, 3)
// ---------------------------------------------------------------------------
__global__ __launch_bounds__(32) void mhsa_qkv_proj(
    const unsigned short* __restrict__ xq, const unsigned short* __restrict__ xk,
    const unsigned short* __restrict__ xv,
    const unsigned short* __restrict__ wqb, const unsigned short* __restrict__ wkb,
    const unsigned short* __restrict__ wvb,
    const float* __restrict__ bq, const float* __restrict__ bk,
    const float* __restrict__ bv,
    unsigned short* __restrict__ oq, unsigned short* __restrict__ ok,
    unsigned short* __restrict__ ov)
{
    const int lane = threadIdx.x & 31;
    const int ln = lane & 15, hi = lane >> 4;
    const int n0 = blockIdx.x * 64;     // output feature tile (64 wide)
    const int m0 = blockIdx.y * 16;     // token tile
    const int which = blockIdx.z;

    const unsigned short* x = which == 0 ? xq : (which == 1 ? xk : xv);
    const unsigned short* w = which == 0 ? wqb : (which == 1 ? wkb : wvb);
    const float* bias       = which == 0 ? bq : (which == 1 ? bk : bv);
    unsigned short* out     = which == 0 ? oq : (which == 1 ? ok : ov);

    v8f acc[4] = {{}, {}, {}, {}};
    for (int k0 = 0; k0 < DIM; k0 += 32) {
        const v16bf a = load_a(x, m0 + ln, k0, hi, DIM);
#pragma unroll
        for (int t = 0; t < 4; ++t) {
            const v16bf b = load_b(w, n0 + t * 16 + ln, k0, hi, DIM); // W^T
            acc[t] = wmma_bf16(a, b, acc[t]);
        }
    }

#pragma unroll
    for (int t = 0; t < 4; ++t) {
        const int col = n0 + t * 16 + ln;
        const float bias_v = bias[col];
        const int h = col >> 6, d = col & 63;
#pragma unroll
        for (int r = 0; r < 8; ++r) {
            const int g = m0 + r + 8 * hi;          // global token row
            const int bb = g >> 11, tk = g & (TOK - 1);
            const unsigned short val = f2bf(acc[t][r] + bias_v);
            if (which == 2)  // V transposed: [B,H,DH,T]
                out[((size_t)(bb * NH + h) * DHEAD + d) * TOK + tk] = val;
            else             // Q/K: [B,H,T,DH]
                out[((size_t)(bb * NH + h) * TOK + tk) * DHEAD + d] = val;
        }
    }
}

// ---------------------------------------------------------------------------
// Kernel 2: flash-style attention, one wave per (b, h, 16-query tile).
// All fragment traffic is b128; P transposed C->A layout through LDS with
// b128 reads. grid = (TOK/16, NH, BATCH)
// ---------------------------------------------------------------------------
__global__ __launch_bounds__(32) void mhsa_flash_attn(
    const unsigned short* __restrict__ Q,
    const unsigned short* __restrict__ K,
    const unsigned short* __restrict__ Vt,   // [B,H,DH,T]
    unsigned short* __restrict__ attn)       // [BT, DIM]
{
    __shared__ unsigned short sP[16 * 32];   // 16 q-rows x 32 key-cols, bf16

    const int lane = threadIdx.x & 31;
    const int ln = lane & 15, hi = lane >> 4;
    const int q0 = blockIdx.x * 16;
    const int h  = blockIdx.y;
    const int b  = blockIdx.z;
    const int bh = b * NH + h;

    const unsigned short* Qp = Q  + (size_t)bh * TOK * DHEAD;
    const unsigned short* Kp = K  + (size_t)bh * TOK * DHEAD;
    const unsigned short* Vp = Vt + (size_t)bh * DHEAD * TOK;

    // Q tile as two A fragments (DH = 64 = 2 x K32)
    const v16bf qa0 = load_a(Qp, q0 + ln, 0,  hi, DHEAD);
    const v16bf qa1 = load_a(Qp, q0 + ln, 32, hi, DHEAD);

    float mrow[8], lrow[8];
    v8f o[4] = {{}, {}, {}, {}};
#pragma unroll
    for (int r = 0; r < 8; ++r) { mrow[r] = -INFINITY; lrow[r] = 0.0f; }

    const float scale = 0.125f;   // 1/sqrt(64)

    for (int s0 = 0; s0 < TOK; s0 += 32) {
        // ---- scores: two 16x16 tiles, 2 WMMAs each over DH ----
        v8f sc[2];
#pragma unroll
        for (int sub = 0; sub < 2; ++sub) {
            const int sb = s0 + sub * 16;
            const v16bf kb0 = load_b(Kp, sb + ln, 0,  hi, DHEAD); // K^T
            const v16bf kb1 = load_b(Kp, sb + ln, 32, hi, DHEAD);
            v8f c = {};
            c = wmma_bf16(qa0, kb0, c);
            c = wmma_bf16(qa1, kb1, c);
            sc[sub] = c;
        }

        // ---- online softmax per row (rows r + 8*hi live in this lane group) ----
#pragma unroll
        for (int r = 0; r < 8; ++r) {
            const float s0v = sc[0][r] * scale;
            const float s1v = sc[1][r] * scale;
            float tmax = fmaxf(s0v, s1v);
            for (int off = 8; off >= 1; off >>= 1)
                tmax = fmaxf(tmax, __shfl_xor(tmax, off, 16));
            const float newm = fmaxf(mrow[r], tmax);
            const float corr = __expf(mrow[r] - newm);
            const float p0 = __expf(s0v - newm);
            const float p1 = __expf(s1v - newm);
            float rs = p0 + p1;
            for (int off = 8; off >= 1; off >>= 1)
                rs += __shfl_xor(rs, off, 16);
            lrow[r] = lrow[r] * corr + rs;
            mrow[r] = newm;
#pragma unroll
            for (int t = 0; t < 4; ++t) o[t][r] *= corr;
            const int row = r + 8 * hi;
            sP[row * 32 + ln]      = f2bf(p0);
            sP[row * 32 + 16 + ln] = f2bf(p1);
        }
        __syncthreads();

        // ---- P back as A fragment: two contiguous b128 LDS reads ----
        Frag pf;
        pf.q[0] = *(const uint4*)(&sP[ln * 32 + 8 * hi]);
        pf.q[1] = *(const uint4*)(&sP[ln * 32 + 16 + 8 * hi]);
        const v16bf pa = pf.v;

        // ---- PV: V^T rows are contiguous along keys -> b128 B fragments ----
#pragma unroll
        for (int t = 0; t < 4; ++t) {
            const v16bf vb = load_b(Vp, t * 16 + ln, s0, hi, TOK);
            o[t] = wmma_bf16(pa, vb, o[t]);
        }
        __syncthreads();
    }

    // ---- normalize + store bf16 rows of the [BT, DIM] attn matrix ----
#pragma unroll
    for (int t = 0; t < 4; ++t) {
#pragma unroll
        for (int r = 0; r < 8; ++r) {
            const int row = q0 + r + 8 * hi;
            const int col = h * DHEAD + t * 16 + ln;
            attn[(size_t)(b * TOK + row) * DIM + col] = f2bf(o[t][r] / lrow[r]);
        }
    }
}

// ---------------------------------------------------------------------------
// Kernel 3: output projection. out = attn @ Wo^T + bo, f32 result.
// One wave per 16x64 tile. grid = (DIM/64, BT/16)
// ---------------------------------------------------------------------------
__global__ __launch_bounds__(32) void mhsa_out_proj(
    const unsigned short* __restrict__ attn,
    const unsigned short* __restrict__ wob,
    const float* __restrict__ bo,
    float* __restrict__ out)
{
    const int lane = threadIdx.x & 31;
    const int ln = lane & 15, hi = lane >> 4;
    const int n0 = blockIdx.x * 64;
    const int m0 = blockIdx.y * 16;

    v8f acc[4] = {{}, {}, {}, {}};
    for (int k0 = 0; k0 < DIM; k0 += 32) {
        const v16bf a = load_a(attn, m0 + ln, k0, hi, DIM);
#pragma unroll
        for (int t = 0; t < 4; ++t) {
            const v16bf b = load_b(wob, n0 + t * 16 + ln, k0, hi, DIM);
            acc[t] = wmma_bf16(a, b, acc[t]);
        }
    }
#pragma unroll
    for (int t = 0; t < 4; ++t) {
        const float bias_v = bo[n0 + t * 16 + ln];
#pragma unroll
        for (int r = 0; r < 8; ++r) {
            out[(size_t)(m0 + r + 8 * hi) * DIM + n0 + t * 16 + ln] = acc[t][r] + bias_v;
        }
    }
}

// ---------------------------------------------------------------------------
extern "C" void kernel_launch(void* const* d_in, const int* in_sizes, int n_in,
                              void* d_out, int out_size, void* d_ws, size_t ws_size,
                              hipStream_t stream) {
    const float* q  = (const float*)d_in[0];
    const float* k  = (const float*)d_in[1];
    const float* v  = (const float*)d_in[2];
    const float* wq = (const float*)d_in[3];
    const float* bq = (const float*)d_in[4];
    const float* wk = (const float*)d_in[5];
    const float* bk = (const float*)d_in[6];
    const float* wv = (const float*)d_in[7];
    const float* bv = (const float*)d_in[8];
    const float* wo = (const float*)d_in[9];
    const float* bo = (const float*)d_in[10];
    float* out = (float*)d_out;

    // Workspace layout (ushorts). attn aliases the x_q bf16 region (dead by
    // the time attention writes it). Peak = 6*E_X + 4*E_W = 56 MB.
    const size_t E_X = (size_t)BT * DIM;    // 4M elements
    const size_t E_W = (size_t)DIM * DIM;   // 1M elements
    unsigned short* ws0   = (unsigned short*)d_ws;
    unsigned short* xq_bf = ws0;
    unsigned short* xk_bf = ws0 + E_X;
    unsigned short* xv_bf = ws0 + 2 * E_X;
    unsigned short* wq_bf = ws0 + 3 * E_X;
    unsigned short* wk_bf = wq_bf + E_W;
    unsigned short* wv_bf = wq_bf + 2 * E_W;
    unsigned short* wo_bf = wq_bf + 3 * E_W;
    unsigned short* Qb    = wq_bf + 4 * E_W;
    unsigned short* Kb    = Qb + E_X;
    unsigned short* Vtb   = Kb + E_X;
    unsigned short* attnb = xq_bf;          // alias

    // 0) f32 -> bf16 conversions (bandwidth-bound, one pass each)
    const int nX = (int)E_X, nW = (int)E_W;
    const int gX = nX / 8 / 256, gW = nW / 8 / 256;
    cvt_f32_to_bf16<<<gX, 256, 0, stream>>>(q,  xq_bf, nX);
    cvt_f32_to_bf16<<<gX, 256, 0, stream>>>(k,  xk_bf, nX);
    cvt_f32_to_bf16<<<gX, 256, 0, stream>>>(v,  xv_bf, nX);
    cvt_f32_to_bf16<<<gW, 256, 0, stream>>>(wq, wq_bf, nW);
    cvt_f32_to_bf16<<<gW, 256, 0, stream>>>(wk, wk_bf, nW);
    cvt_f32_to_bf16<<<gW, 256, 0, stream>>>(wv, wv_bf, nW);
    cvt_f32_to_bf16<<<gW, 256, 0, stream>>>(wo, wo_bf, nW);

    // 1) QKV projection
    dim3 gProj(DIM / 64, BT / 16, 3);        // (16, 256, 3)
    mhsa_qkv_proj<<<gProj, 32, 0, stream>>>(xq_bf, xk_bf, xv_bf,
                                            wq_bf, wk_bf, wv_bf,
                                            bq, bk, bv, Qb, Kb, Vtb);

    // 2) attention
    dim3 gAttn(TOK / 16, NH, BATCH);         // (128, 16, 2)
    mhsa_flash_attn<<<gAttn, 32, 0, stream>>>(Qb, Kb, Vtb, attnb);

    // 3) output projection
    dim3 gOut(DIM / 64, BT / 16);            // (16, 256)
    mhsa_out_proj<<<gOut, 32, 0, stream>>>(attnb, wo_bf, bo, out);
}
